// NodeEdgeCoord_AE_51144470560856
// MI455X (gfx1250) — matrix operand order repaired and checked by reading, not compiled, not run on hardware
//
#include <hip/hip_runtime.h>
#include <hip/hip_bf16.h>
#include <math.h>

#define NN 4096
#define EE 131072
#define IN_NODE 64
#define IN_EDGE 32
#define HID 128
#define OUTF 64
#define EMB 8

// K paddings / strides (in halves)
#define EK 192        // 161 -> 192 (6 k-tiles of 32)
#define ESTR 200      // padded row stride (bank-conflict friendly: 400B = 100 dwords)
#define NK 128        // 99 -> 128 (4 k-tiles)
#define NSTR 136      // 272B = 68 dwords

typedef _Float16 f16;
typedef __attribute__((ext_vector_type(16))) _Float16 v16h;
typedef __attribute__((ext_vector_type(8)))  _Float16 v8h;
typedef __attribute__((ext_vector_type(8)))  float    v8f;
typedef __attribute__((ext_vector_type(2)))  float    v2f;

// ---- output section offsets (flat f32 elements) ----
#define O_NODE_EMB   0
#define O_EDGE_EMB   32768
#define O_RECON_NODE 1081344
#define O_RECON_EDGE 1343488
#define O_ADJ        5537792
#define O_COORD      22315008

// ---- workspace byte offsets ----
#define WS_WE1T   0            // f16 [128][200]  51200 B
#define WS_WE2T   51200       // f16 [32][136]    8704 B
#define WS_WN1T   59904       // f16 [128][136]  34816 B
#define WS_WN2T   94720       // f16 [64][136]   17408 B
#define WS_EOUT   112128      // f32 [E][32]     16 MB
#define WS_AGG    16889344    // f32 [N][32]
#define WS_CACC   17413632    // f32 [N][3]
#define WS_NOUT   17462784    // f32 [N][64]

// A fragment: e[0..7] = K b..b+7, e[8..15] = K b+16..b+23 (per-lane, b = 8*half)
__device__ inline v16h ld_a_frag(const f16* p) {
  union { v16h v; v8h h[2]; } u;
  u.h[0] = *(const v8h*)(p);
  u.h[1] = *(const v8h*)(p + 16);
  return u.v;
}
// B fragment: e[0..7] = K b..b+7, e[8..15] = K b+16..b+23 (per-lane, b = 8*half)
__device__ inline v16h ld_b_frag(const f16* p) {
  union { v16h v; v8h h[2]; } u;
  u.h[0] = *(const v8h*)(p);
  u.h[1] = *(const v8h*)(p + 16);
  return u.v;
}

// ------------------- weight packing (f32 row-major -> f16 transposed panels) ------
__global__ __launch_bounds__(256) void pack_kernel(
    const float* w_e1, const float* w_e2, const float* w_n1, const float* w_n2,
    f16* we1T, f16* we2T, f16* wn1T, f16* wn2T)
{
  int i = blockIdx.x * 256 + threadIdx.x;
  if (i < 128 * EK) {                       // we1T: [161,128] -> [128][ESTR]
    int n = i / EK, k = i % EK;
    we1T[n * ESTR + k] = (f16)(k < 161 ? w_e1[k * 128 + n] : 0.f);
    return;
  }
  i -= 128 * EK;
  if (i < 32 * 128) {                       // we2T: [128,32] -> [32][NSTR]
    int n = i / 128, k = i % 128;
    we2T[n * NSTR + k] = (f16)w_e2[k * 32 + n];
    return;
  }
  i -= 32 * 128;
  if (i < 128 * 128) {                      // wn1T: [99,128] -> [128][NSTR]
    int n = i / 128, k = i % 128;
    wn1T[n * NSTR + k] = (f16)(k < 99 ? w_n1[k * 128 + n] : 0.f);
    return;
  }
  i -= 128 * 128;
  if (i < 64 * 128) {                       // wn2T: [128,64] -> [64][NSTR]
    int n = i / 128, k = i % 128;
    wn2T[n * NSTR + k] = (f16)w_n2[k * 64 + n];
  }
}

// ------------------- init scatter targets ------------------------------------------
__global__ __launch_bounds__(256) void init_kernel(const float* coords, float* cacc, float* agg)
{
  int v = blockIdx.x * 256 + threadIdx.x;
  if (v >= NN) return;
  cacc[v * 3 + 0] = coords[v * 3 + 0];
  cacc[v * 3 + 1] = coords[v * 3 + 1];
  cacc[v * 3 + 2] = coords[v * 3 + 2];
  #pragma unroll
  for (int j = 0; j < 32; ++j) agg[v * 32 + j] = 0.f;
}

// ------------------- edge MLP: gather -> WMMA GEMM1 -> relu -> WMMA GEMM2 ----------
__global__ __launch_bounds__(256) void edge_mlp_kernel(
    const float* feats, const int* eidx, const float* eattr, const float* coords,
    const f16* we1T, const float* be1, const f16* we2T, const float* be2,
    float* edge_out)
{
  __shared__ f16 sBuf[128 * ESTR];
  const int t  = threadIdx.x;
  const int e0 = blockIdx.x * 128;

  // ---- stage edge_in (f16) into LDS: 2 threads per edge ----
  {
    int el = t >> 1, p = t & 1;
    int e  = e0 + el;
    f16* rowp = sBuf + el * ESTR;
    int r = eidx[e];
    if (p == 0) {
      const float4* f4 = (const float4*)(feats + r * IN_NODE);
      #pragma unroll
      for (int j = 0; j < 16; ++j) {
        float4 x = f4[j];
        rowp[4*j+0] = (f16)x.x; rowp[4*j+1] = (f16)x.y;
        rowp[4*j+2] = (f16)x.z; rowp[4*j+3] = (f16)x.w;
      }
    } else {
      int c = eidx[EE + e];
      const float4* f4 = (const float4*)(feats + c * IN_NODE);
      #pragma unroll
      for (int j = 0; j < 16; ++j) {
        float4 x = f4[j];
        rowp[64+4*j+0] = (f16)x.x; rowp[64+4*j+1] = (f16)x.y;
        rowp[64+4*j+2] = (f16)x.z; rowp[64+4*j+3] = (f16)x.w;
      }
      const float4* a4 = (const float4*)(eattr + e * IN_EDGE);
      #pragma unroll
      for (int j = 0; j < 8; ++j) {
        float4 x = a4[j];
        rowp[128+4*j+0] = (f16)x.x; rowp[128+4*j+1] = (f16)x.y;
        rowp[128+4*j+2] = (f16)x.z; rowp[128+4*j+3] = (f16)x.w;
      }
      float dx = coords[r*3+0] - coords[c*3+0];
      float dy = coords[r*3+1] - coords[c*3+1];
      float dz = coords[r*3+2] - coords[c*3+2];
      rowp[160] = (f16)(dx*dx + dy*dy + dz*dz);
      for (int k = 161; k < ESTR; ++k) rowp[k] = (f16)0.f;
    }
  }
  __syncthreads();

  const int lane = t & 31;
  const int wv   = t >> 5;        // 8 waves, one 16-edge M-tile each
  const int l15  = lane & 15;
  const int hf   = lane >> 4;
  const f16* aRow = sBuf + (wv * 16 + l15) * ESTR;

  // ---- GEMM1: [16 x 192] x [192 x 128] ----
  v8f acc[8];
  #pragma unroll
  for (int nt = 0; nt < 8; ++nt) acc[nt] = (v8f){0,0,0,0,0,0,0,0};
  for (int kt = 0; kt < 6; ++kt) {
    v16h a = ld_a_frag(aRow + kt * 32 + hf * 8);
    #pragma unroll
    for (int nt = 0; nt < 8; ++nt) {
      v16h b = ld_b_frag(we1T + (nt * 16 + l15) * ESTR + kt * 32 + hf * 8);
      acc[nt] = __builtin_amdgcn_wmma_f32_16x16x32_f16(
          false, a, false, b, (short)0, acc[nt], false, false);
    }
  }

  // ---- bias + relu, write hidden (f16) back into our own rows of sBuf ----
  #pragma unroll
  for (int nt = 0; nt < 8; ++nt) {
    float bias = be1[nt * 16 + l15];
    #pragma unroll
    for (int r = 0; r < 8; ++r) {
      float x = acc[nt][r] + bias;
      x = x > 0.f ? x : 0.f;
      sBuf[(wv * 16 + r + 8 * hf) * ESTR + nt * 16 + l15] = (f16)x;
    }
  }

  // ---- GEMM2: [16 x 128] x [128 x 32] ----
  v8f acc2[2];
  acc2[0] = (v8f){0,0,0,0,0,0,0,0};
  acc2[1] = (v8f){0,0,0,0,0,0,0,0};
  for (int kt = 0; kt < 4; ++kt) {
    v16h a = ld_a_frag(aRow + kt * 32 + hf * 8);
    #pragma unroll
    for (int nt = 0; nt < 2; ++nt) {
      v16h b = ld_b_frag(we2T + (nt * 16 + l15) * NSTR + kt * 32 + hf * 8);
      acc2[nt] = __builtin_amdgcn_wmma_f32_16x16x32_f16(
          false, a, false, b, (short)0, acc2[nt], false, false);
    }
  }
  #pragma unroll
  for (int nt = 0; nt < 2; ++nt) {
    float bias = be2[nt * 16 + l15];
    #pragma unroll
    for (int r = 0; r < 8; ++r) {
      int e = e0 + wv * 16 + r + 8 * hf;
      edge_out[e * IN_EDGE + nt * 16 + l15] = acc2[nt][r] + bias;
    }
  }
}

// ------------------- per-edge scatter + edge embeddings ----------------------------
__global__ __launch_bounds__(256) void scatter_kernel(
    const float* edge_out, const int* eidx, const float* coords,
    const float* w_ce1, const float* b_ce1, const float* w_ce2,
    const float* w_e2e, const float* b_e2e, const float* w_ee2f, const float* b_ee2f,
    float* agg, float* cacc, float* edge_emb, float* recon_edge)
{
  int e = blockIdx.x * 256 + threadIdx.x;
  float eo[32];
  #pragma unroll
  for (int j = 0; j < 32; ++j) eo[j] = edge_out[e * 32 + j];

  float h0 = b_ce1[0], h1 = b_ce1[1];
  #pragma unroll
  for (int j = 0; j < 32; ++j) { h0 += eo[j] * w_ce1[j*2+0]; h1 += eo[j] * w_ce1[j*2+1]; }
  h0 = h0 > 0.f ? h0 : 0.f;  h1 = h1 > 0.f ? h1 : 0.f;
  float ec = h0 * w_ce2[0] + h1 * w_ce2[1];

  int r = eidx[e], c = eidx[EE + e];
  float dx = coords[r*3+0] - coords[c*3+0];
  float dy = coords[r*3+1] - coords[c*3+1];
  float dz = coords[r*3+2] - coords[c*3+2];
  float rad = dx*dx + dy*dy + dz*dz;
  float inv = ec / (sqrtf(rad) + 1.f);
  atomicAdd(&cacc[r*3+0], dx * inv);
  atomicAdd(&cacc[r*3+1], dy * inv);
  atomicAdd(&cacc[r*3+2], dz * inv);
  #pragma unroll
  for (int j = 0; j < 32; ++j) atomicAdd(&agg[r*32+j], eo[j]);

  float em[8];
  #pragma unroll
  for (int k = 0; k < 8; ++k) em[k] = b_e2e[k];
  #pragma unroll
  for (int j = 0; j < 32; ++j)
    #pragma unroll
    for (int k = 0; k < 8; ++k) em[k] += eo[j] * w_e2e[j*8+k];
  #pragma unroll
  for (int k = 0; k < 8; ++k) edge_emb[e*8+k] = em[k];
  #pragma unroll
  for (int j = 0; j < 32; ++j) {
    float s = b_ee2f[j];
    #pragma unroll
    for (int k = 0; k < 8; ++k) s += em[k] * w_ee2f[k*32+j];
    recon_edge[e*32+j] = s;
  }
}

// ------------------- node MLP ------------------------------------------------------
__global__ __launch_bounds__(256) void node_mlp_kernel(
    const float* feats, const float* agg, const float* cacc,
    const float* w_cm, const float* b_cm,
    const f16* wn1T, const float* bn1, const f16* wn2T, const float* bn2,
    float* coord_out, float* node_out)
{
  __shared__ f16 sBuf[128 * NSTR];
  const int t  = threadIdx.x;
  const int v0 = blockIdx.x * 128;
  {
    int vl = t >> 1, p = t & 1;
    int v  = v0 + vl;
    f16* rowp = sBuf + vl * NSTR;
    if (p == 0) {
      const float4* f4 = (const float4*)(feats + v * IN_NODE);
      #pragma unroll
      for (int j = 0; j < 16; ++j) {
        float4 x = f4[j];
        rowp[4*j+0] = (f16)x.x; rowp[4*j+1] = (f16)x.y;
        rowp[4*j+2] = (f16)x.z; rowp[4*j+3] = (f16)x.w;
      }
    } else {
      #pragma unroll
      for (int j = 0; j < 32; ++j) rowp[64 + j] = (f16)agg[v*32+j];
      float c0 = cacc[v*3+0], c1 = cacc[v*3+1], c2 = cacc[v*3+2];
      #pragma unroll
      for (int j = 0; j < 3; ++j) {
        float co = b_cm[j] + c0*w_cm[0*3+j] + c1*w_cm[1*3+j] + c2*w_cm[2*3+j];
        coord_out[v*3+j] = co;
        rowp[96 + j] = (f16)co;
      }
      for (int k = 99; k < NSTR; ++k) rowp[k] = (f16)0.f;
    }
  }
  __syncthreads();

  const int lane = t & 31;
  const int wv   = t >> 5;
  const int l15  = lane & 15;
  const int hf   = lane >> 4;
  const f16* aRow = sBuf + (wv * 16 + l15) * NSTR;

  v8f acc[8];
  #pragma unroll
  for (int nt = 0; nt < 8; ++nt) acc[nt] = (v8f){0,0,0,0,0,0,0,0};
  for (int kt = 0; kt < 4; ++kt) {
    v16h a = ld_a_frag(aRow + kt * 32 + hf * 8);
    #pragma unroll
    for (int nt = 0; nt < 8; ++nt) {
      v16h b = ld_b_frag(wn1T + (nt * 16 + l15) * NSTR + kt * 32 + hf * 8);
      acc[nt] = __builtin_amdgcn_wmma_f32_16x16x32_f16(
          false, a, false, b, (short)0, acc[nt], false, false);
    }
  }
  #pragma unroll
  for (int nt = 0; nt < 8; ++nt) {
    float bias = bn1[nt * 16 + l15];
    #pragma unroll
    for (int r = 0; r < 8; ++r) {
      float x = acc[nt][r] + bias;
      x = x > 0.f ? x : 0.f;
      sBuf[(wv * 16 + r + 8 * hf) * NSTR + nt * 16 + l15] = (f16)x;
    }
  }

  v8f acc2[4];
  #pragma unroll
  for (int nt = 0; nt < 4; ++nt) acc2[nt] = (v8f){0,0,0,0,0,0,0,0};
  for (int kt = 0; kt < 4; ++kt) {
    v16h a = ld_a_frag(aRow + kt * 32 + hf * 8);
    #pragma unroll
    for (int nt = 0; nt < 4; ++nt) {
      v16h b = ld_b_frag(wn2T + (nt * 16 + l15) * NSTR + kt * 32 + hf * 8);
      acc2[nt] = __builtin_amdgcn_wmma_f32_16x16x32_f16(
          false, a, false, b, (short)0, acc2[nt], false, false);
    }
  }
  #pragma unroll
  for (int nt = 0; nt < 4; ++nt) {
    float bias = bn2[nt * 16 + l15];
    #pragma unroll
    for (int r = 0; r < 8; ++r) {
      int v = v0 + wv * 16 + r + 8 * hf;
      node_out[v * OUTF + nt * 16 + l15] = acc2[nt][r] + bias;
    }
  }
}

// ------------------- node embeddings + reconstruction ------------------------------
__global__ __launch_bounds__(256) void node_head_kernel(
    const float* node_out, const float* w_n2e, const float* b_n2e,
    const float* w_ne2f, const float* b_ne2f, float* node_emb, float* recon_node)
{
  int v = blockIdx.x * 256 + threadIdx.x;
  if (v >= NN) return;
  float em[8];
  #pragma unroll
  for (int k = 0; k < 8; ++k) em[k] = b_n2e[k];
  for (int j = 0; j < OUTF; ++j) {
    float x = node_out[v * OUTF + j];
    #pragma unroll
    for (int k = 0; k < 8; ++k) em[k] += x * w_n2e[j*8+k];
  }
  #pragma unroll
  for (int k = 0; k < 8; ++k) node_emb[v*8+k] = em[k];
  for (int j = 0; j < IN_NODE; ++j) {
    float s = b_ne2f[j];
    #pragma unroll
    for (int k = 0; k < 8; ++k) s += em[k] * w_ne2f[k*IN_NODE+j];
    recon_node[v*IN_NODE+j] = s;
  }
}

// ------------------- adjacency via f32 WMMA Gram matrix ----------------------------
__global__ __launch_bounds__(32) void adj_kernel(
    const float* emb, const float* Wp, const float* bp, float* adj)
{
  const int lane = threadIdx.x;
  const int l15  = lane & 15;
  const int hf   = lane >> 4;
  const int i0   = blockIdx.y * 16;
  const int j0   = blockIdx.x * 16;

  // per-lane norm: lanes 0-15 -> rows of j-tile, lanes 16-31 -> rows of i-tile
  int nrmRow = (lane < 16) ? (j0 + lane) : (i0 + (lane - 16));
  float nrm = 0.f;
  #pragma unroll
  for (int k = 0; k < 8; ++k) { float x = emb[nrmRow*8 + k]; nrm += x * x; }

  v8f g = (v8f){0,0,0,0,0,0,0,0};
  #pragma unroll
  for (int kb = 0; kb < 8; kb += 4) {
    v2f a = *(const v2f*)(emb + (i0 + l15) * 8 + kb + 2 * hf);
    v2f b = *(const v2f*)(emb + (j0 + l15) * 8 + kb + 2 * hf);
    g = __builtin_amdgcn_wmma_f32_16x16x4_f32(
        false, a, false, b, (short)0, g, false, false);
  }

  float W0 = Wp[0], b0 = bp[0];
  float nj = __shfl(nrm, l15, 32);
  #pragma unroll
  for (int r = 0; r < 8; ++r) {
    int m = r + 8 * hf;
    float ni = __shfl(nrm, 16 + m, 32);
    float sq = ni + nj - 2.f * g[r];
    float s  = 1.f / (1.f + __expf(-(W0 * sq + b0)));
    int ig = i0 + m, jg = j0 + l15;
    if (ig == jg) s = 0.f;
    adj[ig * NN + jg] = s;
  }
}

// ===================================================================================
extern "C" void kernel_launch(void* const* d_in, const int* in_sizes, int n_in,
                              void* d_out, int out_size, void* d_ws, size_t ws_size,
                              hipStream_t stream) {
  const float* node_feats = (const float*)d_in[0];
  const int*   edge_index = (const int*)d_in[1];
  const float* edge_attr  = (const float*)d_in[2];
  const float* coords     = (const float*)d_in[3];
  const float* w_e1 = (const float*)d_in[4];  const float* b_e1 = (const float*)d_in[5];
  const float* w_e2 = (const float*)d_in[6];  const float* b_e2 = (const float*)d_in[7];
  const float* w_n1 = (const float*)d_in[8];  const float* b_n1 = (const float*)d_in[9];
  const float* w_n2 = (const float*)d_in[10]; const float* b_n2 = (const float*)d_in[11];
  const float* w_ce1 = (const float*)d_in[12]; const float* b_ce1 = (const float*)d_in[13];
  const float* w_ce2 = (const float*)d_in[14];
  const float* w_cm  = (const float*)d_in[15]; const float* b_cm = (const float*)d_in[16];
  const float* w_n2e = (const float*)d_in[17]; const float* b_n2e = (const float*)d_in[18];
  const float* w_e2e = (const float*)d_in[19]; const float* b_e2e = (const float*)d_in[20];
  const float* w_ne2f = (const float*)d_in[21]; const float* b_ne2f = (const float*)d_in[22];
  const float* w_ee2f = (const float*)d_in[23]; const float* b_ee2f = (const float*)d_in[24];
  const float* Wp = (const float*)d_in[25];    const float* bp = (const float*)d_in[26];

  char* ws = (char*)d_ws;
  f16* we1T = (f16*)(ws + WS_WE1T);
  f16* we2T = (f16*)(ws + WS_WE2T);
  f16* wn1T = (f16*)(ws + WS_WN1T);
  f16* wn2T = (f16*)(ws + WS_WN2T);
  float* edge_out = (float*)(ws + WS_EOUT);
  float* agg      = (float*)(ws + WS_AGG);
  float* cacc     = (float*)(ws + WS_CACC);
  float* node_out = (float*)(ws + WS_NOUT);

  float* out = (float*)d_out;
  float* node_emb   = out + O_NODE_EMB;
  float* edge_emb   = out + O_EDGE_EMB;
  float* recon_node = out + O_RECON_NODE;
  float* recon_edge = out + O_RECON_EDGE;
  float* adj        = out + O_ADJ;
  float* coord_out  = out + O_COORD;

  pack_kernel<<<208, 256, 0, stream>>>(w_e1, w_e2, w_n1, w_n2, we1T, we2T, wn1T, wn2T);
  init_kernel<<<16, 256, 0, stream>>>(coords, cacc, agg);
  edge_mlp_kernel<<<EE / 128, 256, 0, stream>>>(node_feats, edge_index, edge_attr, coords,
                                                we1T, b_e1, we2T, b_e2, edge_out);
  scatter_kernel<<<EE / 256, 256, 0, stream>>>(edge_out, edge_index, coords,
                                               w_ce1, b_ce1, w_ce2, w_e2e, b_e2e,
                                               w_ee2f, b_ee2f, agg, cacc, edge_emb, recon_edge);
  node_mlp_kernel<<<NN / 128, 256, 0, stream>>>(node_feats, agg, cacc, w_cm, b_cm,
                                                wn1T, b_n1, wn2T, b_n2, coord_out, node_out);
  node_head_kernel<<<NN / 256, 256, 0, stream>>>(node_out, w_n2e, b_n2e, w_ne2f, b_ne2f,
                                                 node_emb, recon_node);
  adj_kernel<<<dim3(NN / 16, NN / 16, 1), 32, 0, stream>>>(node_emb, Wp, bp, adj);
}
